// ProbAttention_755914244461
// MI455X (gfx1250) — matrix operand order repaired and checked by reading, not compiled, not run on hardware
//
#include <hip/hip_runtime.h>
#include <math.h>

// ---------------- types ----------------
typedef __attribute__((ext_vector_type(16))) __bf16 v16bf;
typedef __attribute__((ext_vector_type(8)))  __bf16 v8bf;
typedef __attribute__((ext_vector_type(8)))  float  v8f;

#define B_   16
#define L_   2048
#define DIN_ 512
#define HID_ 512
#define H_   8
#define D_   64
#define U_   40
#define UP_  48
#define BH_  (B_ * H_)
#define MTOT (B_ * L_)   // 32768

__device__ __forceinline__ unsigned pk2(float a, float b) {
  union { __bf16 h[2]; unsigned u; } x;
  x.h[0] = (__bf16)a; x.h[1] = (__bf16)b;
  return x.u;
}

// ---- CDNA5 async global->LDS copy (ASYNCcnt path) ----
#if __has_builtin(__builtin_amdgcn_global_load_async_to_lds_b128)
#define ASYNC_BUILTIN 1
#endif

typedef int v4i_ __attribute__((vector_size(16)));
typedef v4i_ __attribute__((address_space(1)))* as1_v4i_p;
typedef v4i_ __attribute__((address_space(3)))* as3_v4i_p;

__device__ __forceinline__ void async_copy_b128(const void* g, void* lds) {
#if defined(ASYNC_BUILTIN)
  __builtin_amdgcn_global_load_async_to_lds_b128((as1_v4i_p)g, (as3_v4i_p)lds, 0, 0);
#else
  unsigned loff = (unsigned)(size_t)lds;
  asm volatile("global_load_async_to_lds_b128 %0, %1, off"
               :: "v"(loff), "v"(g) : "memory");
#endif
}

__device__ __forceinline__ void async_wait0() {
#if __has_builtin(__builtin_amdgcn_s_wait_asynccnt)
  __builtin_amdgcn_s_wait_asynccnt(0);
#else
  asm volatile("s_wait_asynccnt 0" ::: "memory");
#endif
}

// ---- CDNA5 Tensor Data Mover (TENSORcnt path) ----
#if __has_builtin(__builtin_amdgcn_tensor_load_to_lds)
#define HAVE_TDM 1
#endif

typedef unsigned int u32x4_t __attribute__((ext_vector_type(4)));
typedef int i32x4_t __attribute__((ext_vector_type(4)));
typedef int i32x8_t __attribute__((ext_vector_type(8)));

__device__ __forceinline__ void tensor_wait0() {
#if __has_builtin(__builtin_amdgcn_s_wait_tensorcnt)
  __builtin_amdgcn_s_wait_tensorcnt(0);
#else
  asm volatile("s_wait_tensorcnt 0" ::: "memory");
#endif
}

// Load a 2-D tile (tile_w x tile_h elements, 2-byte elements) from a
// row-major tensor into LDS; pad 4 DWORDs after every 16 DWORDs stored
// (i.e. 64B data rows -> 80B LDS row stride, matching [..][40] bf16 arrays).
__device__ __forceinline__ void tdm_load_2d_bf16(
    unsigned lds_addr, unsigned long long gaddr,
    unsigned tensor_w, unsigned tensor_h,
    unsigned tile_w, unsigned tile_h, unsigned row_stride)
{
#if defined(HAVE_TDM)
  u32x4_t g0;
  g0[0] = 1u;                                   // count=1, user mode
  g0[1] = lds_addr;                             // LDS byte address
  g0[2] = (unsigned)(gaddr & 0xffffffffu);      // global_addr[31:0]
  g0[3] = (unsigned)((gaddr >> 32) & 0x1ffffffu) | (2u << 30);  // [56:32] | type=2
  i32x8_t g1;
  g1[0] = (int)((1u << 16) |                    // data_size = 2 bytes
                (1u << 20) |                    // pad_enable
                (3u << 22) |                    // pad_interval: 16 DWORDs
                (3u << 25));                    // pad_amount:   4 DWORDs
  g1[1] = (int)((tensor_w & 0xffffu) << 16);                    // tensor_dim0[15:0]
  g1[2] = (int)((tensor_w >> 16) | ((tensor_h & 0xffffu) << 16)); // dim0[31:16], dim1[15:0]
  g1[3] = (int)((tensor_h >> 16) | (tile_w << 16));             // dim1[31:16], tile_dim0
  g1[4] = (int)(tile_h & 0xffffu);                              // tile_dim1 (tile_dim2=0)
  g1[5] = (int)row_stride;                                      // tensor_dim0_stride[31:0]
  g1[6] = 0;                                                    // stride[47:32], dim1_stride lo
  g1[7] = 0;
  i32x4_t z4 = {0, 0, 0, 0};
  i32x8_t z8 = {0, 0, 0, 0, 0, 0, 0, 0};
  __builtin_amdgcn_tensor_load_to_lds(g0, g1, z4, z4, z8, 0);
#else
  (void)lds_addr; (void)gaddr; (void)tensor_w; (void)tensor_h;
  (void)tile_w; (void)tile_h; (void)row_stride;
#endif
}

// A-fragment (16x32 bf16): lane (g = lane>>4, m = lane&15)
// elems 0..15 = row[g*16 + j]
__device__ __forceinline__ v16bf ld_a_frag(const __bf16* row, int g) {
  v8bf lo = *reinterpret_cast<const v8bf*>(row + g * 16);
  v8bf hi = *reinterpret_cast<const v8bf*>(row + g * 16 + 8);
  return __builtin_shufflevector(lo, hi, 0,1,2,3,4,5,6,7,8,9,10,11,12,13,14,15);
}

// B-fragment (32x16 bf16): lane (g, n): 16 contiguous K values of column n
__device__ __forceinline__ v16bf ld_b_frag(const __bf16* p) {
  v8bf lo = *reinterpret_cast<const v8bf*>(p);
  v8bf hi = *reinterpret_cast<const v8bf*>(p + 8);
  return __builtin_shufflevector(lo, hi, 0,1,2,3,4,5,6,7,8,9,10,11,12,13,14,15);
}

__device__ __forceinline__ v8f wmma_bf16(v16bf a, v16bf b, v8f c) {
  return __builtin_amdgcn_wmma_f32_16x16x32_bf16(false, a, false, b, (short)0, c, false, false);
}

// ============================================================
// Kernel 1: projections Q/K/V = X @ W (f32 in, bf16 out)
// ============================================================
__global__ __launch_bounds__(256) void proj_gemm_kernel(
    const float* __restrict__ q_in, const float* __restrict__ k_in, const float* __restrict__ v_in,
    const float* __restrict__ Wq,  const float* __restrict__ Wk,  const float* __restrict__ Wv,
    __bf16* __restrict__ Qo, __bf16* __restrict__ Ko, __bf16* __restrict__ Vo)
{
  const float* X; const float* W; __bf16* O;
  if (blockIdx.z == 0)      { X = q_in; W = Wq; O = Qo; }
  else if (blockIdx.z == 1) { X = k_in; W = Wk; O = Ko; }
  else                      { X = v_in; W = Wv; O = Vo; }

  __shared__ alignas(16) __bf16 As[128][40];
  __shared__ alignas(16) __bf16 Bs[128][40];

  const int tid  = threadIdx.x;
  const int wave = tid >> 5, lane = tid & 31;
  const int wm = wave >> 2, wn = wave & 3;
  const int g  = lane >> 4, ln = lane & 15;
  const int m0 = blockIdx.x * 128;
  const int n0 = blockIdx.y * 128;

  v8f acc[4][2];
  for (int i = 0; i < 4; ++i)
    for (int j = 0; j < 2; ++j)
      for (int r = 0; r < 8; ++r) acc[i][j][r] = 0.f;

  for (int k0 = 0; k0 < DIN_; k0 += 32) {
    #pragma unroll
    for (int s = 0; s < 4; ++s) {
      int idx = tid + s * 256;
      int row = idx >> 3, kq = (idx & 7) << 2;
      float4 f = *reinterpret_cast<const float4*>(&X[(size_t)(m0 + row) * DIN_ + k0 + kq]);
      uint2 p; p.x = pk2(f.x, f.y); p.y = pk2(f.z, f.w);
      *reinterpret_cast<uint2*>(&As[row][kq]) = p;
    }
    #pragma unroll
    for (int s = 0; s < 4; ++s) {
      int idx = tid + s * 256;
      int kr = idx >> 5, nq = (idx & 31) << 2;
      float4 f = *reinterpret_cast<const float4*>(&W[(size_t)(k0 + kr) * HID_ + n0 + nq]);
      Bs[nq + 0][kr] = (__bf16)f.x;
      Bs[nq + 1][kr] = (__bf16)f.y;
      Bs[nq + 2][kr] = (__bf16)f.z;
      Bs[nq + 3][kr] = (__bf16)f.w;
    }
    __syncthreads();

    v16bf a[4], b[2];
    #pragma unroll
    for (int i = 0; i < 4; ++i) a[i] = ld_a_frag(&As[wm * 64 + i * 16 + ln][0], g);
    #pragma unroll
    for (int j = 0; j < 2; ++j) b[j] = ld_b_frag(&Bs[wn * 32 + j * 16 + ln][g * 16]);
    #pragma unroll
    for (int i = 0; i < 4; ++i)
      #pragma unroll
      for (int j = 0; j < 2; ++j)
        acc[i][j] = wmma_bf16(a[i], b[j], acc[i][j]);
    __syncthreads();
  }

  #pragma unroll
  for (int i = 0; i < 4; ++i)
    for (int j = 0; j < 2; ++j) {
      int col  = n0 + wn * 32 + j * 16 + ln;
      int rowb = m0 + wm * 64 + i * 16 + g * 8;
      #pragma unroll
      for (int r = 0; r < 8; ++r)
        O[(size_t)(rowb + r) * HID_ + col] = (__bf16)acc[i][j][r];
    }
}

// ============================================================
// Kernel 2: sampled dot + top-U selection; one WG per (b,h)
// ============================================================
__global__ __launch_bounds__(256) void dot_topk_kernel(
    const __bf16* __restrict__ Q, const __bf16* __restrict__ K,
    const int* __restrict__ indx, int* __restrict__ mtop)
{
  const int bh = blockIdx.x;
  const size_t base = (size_t)bh * L_ * D_;
  __shared__ float Ms[L_];
  __shared__ float rv[256];
  __shared__ int   ri[256];
  const float scl = 1.0f - (float)U_ / (float)L_;

  for (int l = threadIdx.x; l < L_; l += 256) {
    int kr = indx[l];
    const v8bf* qp = reinterpret_cast<const v8bf*>(&Q[base + (size_t)l * D_]);
    const v8bf* kp = reinterpret_cast<const v8bf*>(&K[base + (size_t)kr * D_]);
    float s = 0.f;
    #pragma unroll
    for (int c = 0; c < 8; ++c) {
      v8bf a = qp[c], b = kp[c];
      #pragma unroll
      for (int e = 0; e < 8; ++e) s += (float)a[e] * (float)b[e];
    }
    Ms[l] = s * scl;
  }
  __syncthreads();

  for (int u = 0; u < U_; ++u) {
    float bv = -3.0e38f; int bi = 0x7fffffff;
    for (int l = threadIdx.x; l < L_; l += 256) {
      float v = Ms[l];
      if (v > bv) { bv = v; bi = l; }
    }
    rv[threadIdx.x] = bv; ri[threadIdx.x] = bi;
    __syncthreads();
    for (int off = 128; off > 0; off >>= 1) {
      if (threadIdx.x < off) {
        float ov = rv[threadIdx.x + off];
        int   oi = ri[threadIdx.x + off];
        if (ov > rv[threadIdx.x] || (ov == rv[threadIdx.x] && oi < ri[threadIdx.x])) {
          rv[threadIdx.x] = ov; ri[threadIdx.x] = oi;
        }
      }
      __syncthreads();
    }
    if (threadIdx.x == 0) { int w = ri[0]; mtop[bh * U_ + u] = w; Ms[w] = -3.0e38f; }
    __syncthreads();
  }
}

// ============================================================
// Kernel 3: scores /8, causal mask, softmax, attn@V via WMMA.
// ============================================================
__global__ __launch_bounds__(256) void attn_kernel(
    const __bf16* __restrict__ Q, const __bf16* __restrict__ K, const __bf16* __restrict__ V,
    const int* __restrict__ mtop, float* __restrict__ upd)
{
  const int bh = blockIdx.x;
  const size_t base = (size_t)bh * L_ * D_;
  const int tid = threadIdx.x, wave = tid >> 5, lane = tid & 31;
  const int g = lane >> 4, ln = lane & 15;

  __shared__ alignas(16) __bf16 qred[UP_][72];
  __shared__ alignas(16) __bf16 esc[8][UP_][40];
  __shared__ alignas(16) __bf16 vlds[8][32][72];
  __shared__ float updacc[UP_][64];
  __shared__ float denomS[UP_];
  __shared__ float rowmaxS[UP_];
  __shared__ float wpmax[8][UP_];
  __shared__ int   msel[UP_];

  for (int idx = tid; idx < UP_ * 8; idx += 256) {
    int u = idx >> 3, c = (idx & 7) * 8;
    if (u < U_) {
      int m = mtop[bh * U_ + u];
      *reinterpret_cast<uint4*>(&qred[u][c]) =
          *reinterpret_cast<const uint4*>(&Q[base + (size_t)m * D_ + c]);
    } else {
      uint4 z; z.x = z.y = z.z = z.w = 0u;
      *reinterpret_cast<uint4*>(&qred[u][c]) = z;
    }
  }
  for (int idx = tid; idx < UP_ * 64; idx += 256) {
    int u = idx >> 6, d = idx & 63;
    updacc[u][d] = 0.f;
  }
  if (tid < UP_) {
    msel[tid]  = (tid < U_) ? mtop[bh * U_ + tid] : (1 << 30);
    denomS[tid] = 0.f;
  }
  __syncthreads();

  const int l0w = wave * 256;

  // ---- pass A: row maxima ----
  float rm[3][8];
  for (int mt = 0; mt < 3; ++mt)
    for (int r = 0; r < 8; ++r) rm[mt][r] = -3.0e38f;

  for (int t = 0; t < 16; ++t) {
    int lt = l0w + t * 16;
    int lrow = lt + ln;
    v16bf bfr[2];
    #pragma unroll
    for (int kc = 0; kc < 2; ++kc)
      bfr[kc] = ld_b_frag(&K[base + (size_t)lrow * D_ + kc * 32 + g * 16]);
    #pragma unroll
    for (int mt = 0; mt < 3; ++mt) {
      v8f acc; for (int r = 0; r < 8; ++r) acc[r] = 0.f;
      #pragma unroll
      for (int kc = 0; kc < 2; ++kc)
        acc = wmma_bf16(ld_a_frag(&qred[mt * 16 + ln][kc * 32], g), bfr[kc], acc);
      int col = lt + ln;
      #pragma unroll
      for (int r = 0; r < 8; ++r) {
        int row = mt * 16 + g * 8 + r;
        float s = acc[r] * 0.125f;
        if (col > msel[row]) s = -3.0e38f;
        rm[mt][r] = fmaxf(rm[mt][r], s);
      }
    }
  }
  for (int mt = 0; mt < 3; ++mt)
    for (int r = 0; r < 8; ++r) {
      float v = rm[mt][r];
      v = fmaxf(v, __shfl_xor(v, 1));
      v = fmaxf(v, __shfl_xor(v, 2));
      v = fmaxf(v, __shfl_xor(v, 4));
      v = fmaxf(v, __shfl_xor(v, 8));
      rm[mt][r] = v;
    }
  if (ln == 0)
    for (int mt = 0; mt < 3; ++mt)
      for (int r = 0; r < 8; ++r) wpmax[wave][mt * 16 + g * 8 + r] = rm[mt][r];
  __syncthreads();
  if (tid < UP_) {
    float m = -3.0e38f;
    for (int w = 0; w < 8; ++w) m = fmaxf(m, wpmax[w][tid]);
    rowmaxS[tid] = m;
  }
  __syncthreads();

  // ---- pass B: exp + denom + attn@V (WMMA) ----
  v8f uacc[3][4];
  for (int mt = 0; mt < 3; ++mt)
    for (int dt = 0; dt < 4; ++dt)
      for (int r = 0; r < 8; ++r) uacc[mt][dt][r] = 0.f;
  float dsum[3][8];
  for (int mt = 0; mt < 3; ++mt)
    for (int r = 0; r < 8; ++r) dsum[mt][r] = 0.f;

  for (int t2 = 0; t2 < 8; ++t2) {
    int lc = l0w + t2 * 32;

    #pragma unroll
    for (int s = 0; s < 8; ++s) {
      int idx = lane + s * 32;
      int rr = idx >> 3, cc = (idx & 7) * 8;
      async_copy_b128(&V[base + (size_t)(lc + rr) * D_ + cc], &vlds[wave][rr][cc]);
    }

    #pragma unroll
    for (int ns = 0; ns < 2; ++ns) {
      int lt = lc + ns * 16;
      int lrow = lt + ln;
      v16bf bfr[2];
      #pragma unroll
      for (int kc = 0; kc < 2; ++kc)
        bfr[kc] = ld_b_frag(&K[base + (size_t)lrow * D_ + kc * 32 + g * 16]);
      #pragma unroll
      for (int mt = 0; mt < 3; ++mt) {
        v8f acc; for (int r = 0; r < 8; ++r) acc[r] = 0.f;
        #pragma unroll
        for (int kc = 0; kc < 2; ++kc)
          acc = wmma_bf16(ld_a_frag(&qred[mt * 16 + ln][kc * 32], g), bfr[kc], acc);
        int col = lt + ln;
        #pragma unroll
        for (int r = 0; r < 8; ++r) {
          int row = mt * 16 + g * 8 + r;
          float s = acc[r] * 0.125f;
          if (col > msel[row]) s = -3.0e38f;
          float p = __expf(s - rowmaxS[row]);
          dsum[mt][r] += p;
          esc[wave][row][ns * 16 + ln] = (__bf16)p;
        }
      }
    }

    async_wait0();

    #pragma unroll
    for (int dt = 0; dt < 4; ++dt) {
      v16bf vfr;
      #pragma unroll
      for (int e = 0; e < 16; ++e)
        vfr[e] = vlds[wave][g * 16 + e][dt * 16 + ln];
      #pragma unroll
      for (int mt = 0; mt < 3; ++mt)
        uacc[mt][dt] = wmma_bf16(ld_a_frag(&esc[wave][mt * 16 + ln][0], g), vfr, uacc[mt][dt]);
    }
  }

  for (int mt = 0; mt < 3; ++mt)
    for (int r = 0; r < 8; ++r) {
      float v = dsum[mt][r];
      v += __shfl_xor(v, 1);
      v += __shfl_xor(v, 2);
      v += __shfl_xor(v, 4);
      v += __shfl_xor(v, 8);
      if (ln == 0) atomicAdd(&denomS[mt * 16 + g * 8 + r], v);
    }
  for (int mt = 0; mt < 3; ++mt)
    for (int dt = 0; dt < 4; ++dt)
      for (int r = 0; r < 8; ++r)
        atomicAdd(&updacc[mt * 16 + g * 8 + r][dt * 16 + ln], uacc[mt][dt][r]);
  __syncthreads();

  for (int idx = tid; idx < U_ * 64; idx += 256) {
    int u = idx >> 6, d = idx & 63;
    upd[((size_t)bh * U_ + u) * 64 + d] = updacc[u][d] / denomS[u];
  }
}

// ============================================================
// Kernel 4: ctx = cumsum(V) along L (f32 accumulate, bf16 store)
// ============================================================
__global__ __launch_bounds__(256) void cumsum_kernel(
    const __bf16* __restrict__ V, __bf16* __restrict__ ctx)
{
  const int bh = blockIdx.x;
  const size_t base = (size_t)bh * L_ * D_;
  const int d = threadIdx.x & 63;
  const int c = threadIdx.x >> 6;
  __shared__ float csum[4][64];

  float acc = 0.f;
  for (int l = c * 512; l < (c + 1) * 512; ++l) acc += (float)V[base + (size_t)l * D_ + d];
  csum[c][d] = acc;
  __syncthreads();
  float run = 0.f;
  for (int cc = 0; cc < c; ++cc) run += csum[cc][d];
  for (int l = c * 512; l < (c + 1) * 512; ++l) {
    run += (float)V[base + (size_t)l * D_ + d];
    ctx[base + (size_t)l * D_ + d] = (__bf16)run;
  }
}

// ============================================================
// Kernel 5: scatter upd rows into ctx at m_top
// ============================================================
__global__ __launch_bounds__(256) void scatter_kernel(
    const int* __restrict__ mtop, const float* __restrict__ upd, __bf16* __restrict__ ctx)
{
  const int bh = blockIdx.x;
  for (int idx = threadIdx.x; idx < U_ * 64; idx += 256) {
    int u = idx >> 6, d = idx & 63;
    int m = mtop[bh * U_ + u];
    ctx[(size_t)bh * L_ * D_ + (size_t)m * D_ + d] = (__bf16)upd[((size_t)bh * U_ + u) * 64 + d];
  }
}

// ============================================================
// Kernel 6: out = ctx(32768x512 bf16) @ Wo(512x64 f32) -> f32
// A tile DMA'd by the Tensor Data Mover (TENSORcnt), with
// async-to-LDS fallback if the TDM builtin is unavailable.
// ============================================================
__global__ __launch_bounds__(256) void out_gemm_kernel(
    const __bf16* __restrict__ Xc, const float* __restrict__ Wo, float* __restrict__ Out)
{
  __shared__ alignas(16) __bf16 As[128][40];
  __shared__ alignas(16) __bf16 Bs[64][40];
  const int tid = threadIdx.x, wave = tid >> 5, lane = tid & 31;
  const int wm = wave >> 1, wn = wave & 1;
  const int g = lane >> 4, ln = lane & 15;
  const int m0 = blockIdx.x * 128;

  v8f acc[2][2];
  for (int i = 0; i < 2; ++i)
    for (int j = 0; j < 2; ++j)
      for (int r = 0; r < 8; ++r) acc[i][j][r] = 0.f;

  for (int k0 = 0; k0 < HID_; k0 += 32) {
#if defined(HAVE_TDM)
    // one wave issues the 128x32 bf16 tile DMA, waits TENSORcnt
    if (wave == 0) {
      tdm_load_2d_bf16((unsigned)(size_t)&As[0][0],
                       (unsigned long long)(size_t)&Xc[(size_t)m0 * HID_ + k0],
                       /*tensor_w=*/HID_, /*tensor_h=*/MTOT,
                       /*tile_w=*/32, /*tile_h=*/128, /*row_stride=*/HID_);
      tensor_wait0();
    }
#else
    #pragma unroll
    for (int s = 0; s < 2; ++s) {
      int idx = tid + s * 256;
      int row = idx >> 2, cq = (idx & 3) * 8;
      async_copy_b128(&Xc[(size_t)(m0 + row) * HID_ + k0 + cq], &As[row][cq]);
    }
#endif
    // stage B tile (f32 -> bf16, transposed [n][k])
    #pragma unroll
    for (int s = 0; s < 2; ++s) {
      int idx = tid + s * 256;
      int kr = idx >> 4, nq = (idx & 15) << 2;
      float4 f = *reinterpret_cast<const float4*>(&Wo[(size_t)(k0 + kr) * 64 + nq]);
      Bs[nq + 0][kr] = (__bf16)f.x;
      Bs[nq + 1][kr] = (__bf16)f.y;
      Bs[nq + 2][kr] = (__bf16)f.z;
      Bs[nq + 3][kr] = (__bf16)f.w;
    }
#if !defined(HAVE_TDM)
    async_wait0();
#endif
    __syncthreads();

    v16bf a[2], b[2];
    #pragma unroll
    for (int i = 0; i < 2; ++i) a[i] = ld_a_frag(&As[wm * 32 + i * 16 + ln][0], g);
    #pragma unroll
    for (int j = 0; j < 2; ++j) b[j] = ld_b_frag(&Bs[wn * 32 + j * 16 + ln][g * 16]);
    #pragma unroll
    for (int i = 0; i < 2; ++i)
      #pragma unroll
      for (int j = 0; j < 2; ++j)
        acc[i][j] = wmma_bf16(a[i], b[j], acc[i][j]);
    __syncthreads();
  }

  #pragma unroll
  for (int i = 0; i < 2; ++i)
    for (int j = 0; j < 2; ++j) {
      int col  = wn * 32 + j * 16 + ln;
      int rowb = m0 + wm * 32 + i * 16 + g * 8;
      #pragma unroll
      for (int r = 0; r < 8; ++r)
        Out[(size_t)(rowb + r) * 64 + col] = acc[i][j][r];
    }
}

// ============================================================
extern "C" void kernel_launch(void* const* d_in, const int* in_sizes, int n_in,
                              void* d_out, int out_size, void* d_ws, size_t ws_size,
                              hipStream_t stream)
{
  const float* q_in = (const float*)d_in[0];
  const float* k_in = (const float*)d_in[1];
  const float* v_in = (const float*)d_in[2];
  const float* Wq   = (const float*)d_in[3];
  const float* Wk   = (const float*)d_in[4];
  const float* Wv   = (const float*)d_in[5];
  const float* Wo   = (const float*)d_in[6];
  const int*   indx = (const int*)d_in[7];
  float* out = (float*)d_out;

  char* ws = (char*)d_ws;
  const size_t SZB = (size_t)MTOT * HID_ * sizeof(__bf16);   // 32 MB each
  __bf16* Qb   = (__bf16*)(ws);
  __bf16* Kb   = (__bf16*)(ws + SZB);
  __bf16* Vb   = (__bf16*)(ws + 2 * SZB);
  __bf16* ctxb = (__bf16*)(ws + 3 * SZB);
  int*    mtop = (int*)   (ws + 4 * SZB);
  float*  upd  = (float*) (ws + 4 * SZB + ((size_t)(BH_ * U_ * sizeof(int) + 255) & ~(size_t)255));

  proj_gemm_kernel<<<dim3(MTOT / 128, HID_ / 128, 3), 256, 0, stream>>>(
      q_in, k_in, v_in, Wq, Wk, Wv, Qb, Kb, Vb);
  dot_topk_kernel<<<dim3(BH_), 256, 0, stream>>>(Qb, Kb, indx, mtop);
  attn_kernel<<<dim3(BH_), 256, 0, stream>>>(Qb, Kb, Vb, mtop, upd);
  cumsum_kernel<<<dim3(BH_), 256, 0, stream>>>(Vb, ctxb);
  scatter_kernel<<<dim3(BH_), 256, 0, stream>>>(mtop, upd, ctxb);
  out_gemm_kernel<<<dim3(MTOT / 128, 1), 256, 0, stream>>>(ctxb, Wo, out);
}
